// MindSpeedTEGroupedLinear_44856638439908
// MI455X (gfx1250) — compile-verified, matrix operand order checked by reading, not run on hardware
//
#include <hip/hip_runtime.h>

// ---------------------------------------------------------------------------
// MindSpeed TE grouped linear: out[m, n] = sum_k x[m, k] * W[e(m), n, k]
// fp32 in / fp32 out, internally bf16 WMMA (v_wmma_f32_16x16x32_bf16).
// Block tile 128x256, 8 waves, each wave computes a 64x64 sub-tile
// (4x4 grid of 16x16 WMMA accumulators) -> 1 ds_load_b128 per WMMA.
// ---------------------------------------------------------------------------

typedef __bf16 bf16_t;
typedef __attribute__((ext_vector_type(16))) __bf16 v16bf;
typedef __attribute__((ext_vector_type(8)))  __bf16 bf16x8;
typedef __attribute__((ext_vector_type(4)))  __bf16 bf16x4;
typedef __attribute__((ext_vector_type(8)))  float  v8f;

namespace cfg {
constexpr int kDIn  = 2048;
constexpr int kDOut = 8192;
constexpr int kT    = 16384;
constexpr int kBM   = 128;
constexpr int kBN   = 256;
constexpr int kBK   = 64;
constexpr int kLdk  = kBK + 8;  // padded LDS stride (bf16) -> conflict-free b128 reads
}

__global__ __launch_bounds__(256, 2)
void grouped_gemm_bf16_wmma(const float* __restrict__ x,
                            const float* __restrict__ w,
                            float* __restrict__ out) {
    using namespace cfg;

    __shared__ bf16_t sA[kBM][kLdk];   // x tile   (rows = M)       18 KB
    __shared__ bf16_t sB[kBN][kLdk];   // W tile   (rows = N = W_e row) 36 KB

    const int m0 = blockIdx.y * kBM;
    const int n0 = blockIdx.x * kBN;

    // Cumsum of M_SPLITS; all boundaries are multiples of 128, so a 128-row
    // M tile never straddles experts.
    const int offs[9] = {0, 1024, 4096, 6144, 8192, 9216, 12288, 14848, 16384};
    int e = 0;
#pragma unroll
    for (int i = 1; i < 8; ++i)
        if (m0 >= offs[i]) e = i;

    // Scalar (SGPR) bases + 32-bit per-lane offsets -> saddr-form global ops.
    const float* __restrict__ xb = x + (size_t)m0 * kDIn;
    const float* __restrict__ wb = w + (size_t)e * (size_t)kDOut * (size_t)kDIn
                                     + (size_t)n0 * kDIn;
    float* __restrict__ ob = out + (size_t)m0 * kDOut + n0;

    const int tid  = threadIdx.x;
    const int lane = tid & 31;
    const int wave = tid >> 5;
    const int wm   = wave >> 2;             // 0..1 -> 64-row band
    const int wn   = wave & 3;              // 0..3 -> 64-col band
    const int hgrp = lane >> 4;             // lane half (0/1)
    const int lrow = lane & 15;             // row/col within 16

    v8f acc[4][4];
#pragma unroll
    for (int i = 0; i < 4; ++i)
#pragma unroll
        for (int j = 0; j < 4; ++j)
            acc[i][j] = (v8f){0.f, 0.f, 0.f, 0.f, 0.f, 0.f, 0.f, 0.f};

    for (int k0 = 0; k0 < kDIn; k0 += kBK) {
        // ---- global->LDS stage with fp32 -> bf16 conversion ----
        // sA: 128x64 fp32 = 2048 float4, 8 per thread.
#pragma unroll
        for (int r = 0; r < 8; ++r) {
            const unsigned f   = (unsigned)tid + r * 256u;   // 0..2047
            const unsigned row = f >> 4;                     // 0..127
            const unsigned col = (f & 15u) << 2;             // 0..60
            const float4 v = *(const float4*)(xb + row * (unsigned)kDIn
                                                 + (unsigned)k0 + col);
            const bf16x4 b = { (bf16_t)v.x, (bf16_t)v.y, (bf16_t)v.z, (bf16_t)v.w };
            *(bf16x4*)&sA[row][col] = b;
        }
        // sB: 256x64 fp32 = 4096 float4, 16 per thread.
#pragma unroll
        for (int r = 0; r < 16; ++r) {
            const unsigned f   = (unsigned)tid + r * 256u;   // 0..4095
            const unsigned row = f >> 4;                     // 0..255
            const unsigned col = (f & 15u) << 2;
            const float4 v = *(const float4*)(wb + row * (unsigned)kDIn
                                                 + (unsigned)k0 + col);
            const bf16x4 b = { (bf16_t)v.x, (bf16_t)v.y, (bf16_t)v.z, (bf16_t)v.w };
            *(bf16x4*)&sB[row][col] = b;
        }

        // Prefetch next K chunk while this one is consumed.
        if (k0 + kBK < kDIn) {
            __builtin_prefetch(xb + (unsigned)((tid >> 1) * kDIn + k0 + kBK + (tid & 1) * 32), 0, 1);
            __builtin_prefetch(wb + (unsigned)(tid * kDIn + k0 + kBK), 0, 1);
        }

        __syncthreads();

        // ---- WMMA over the staged K chunk (two k-steps of 32) ----
#pragma unroll
        for (int kk = 0; kk < kBK; kk += 32) {
            const int klo = kk + hgrp * 8;   // 16-bit A/B layout: lane half owns K+8

            v16bf afrag[4];
#pragma unroll
            for (int i = 0; i < 4; ++i) {
                const int row = wm * 64 + i * 16 + lrow;
                const bf16x8 lo = *(const bf16x8*)&sA[row][klo];
                const bf16x8 hi = *(const bf16x8*)&sA[row][klo + 16];
                v16bf a;
#pragma unroll
                for (int t = 0; t < 8; ++t) { a[t] = lo[t]; a[t + 8] = hi[t]; }
                afrag[i] = a;
            }

            v16bf bfrag[4];
#pragma unroll
            for (int j = 0; j < 4; ++j) {
                const int row = wn * 64 + j * 16 + lrow;   // W_e row == output col
                const bf16x8 lo = *(const bf16x8*)&sB[row][klo];
                const bf16x8 hi = *(const bf16x8*)&sB[row][klo + 16];
                v16bf b;
#pragma unroll
                for (int t = 0; t < 8; ++t) { b[t] = lo[t]; b[t + 8] = hi[t]; }
                bfrag[j] = b;
            }

#pragma unroll
            for (int i = 0; i < 4; ++i)
#pragma unroll
                for (int j = 0; j < 4; ++j)
                    acc[i][j] = __builtin_amdgcn_wmma_f32_16x16x32_bf16(
                        /*neg_a=*/false, afrag[i],
                        /*neg_b=*/false, bfrag[j],
                        /*c_mod=*/(short)0, acc[i][j],
                        /*reuse_a=*/false, /*reuse_b=*/false);
        }

        __syncthreads();
    }

    // ---- store: C/D layout -> VGPR r is row r, lane half owns M+8 ----
#pragma unroll
    for (int i = 0; i < 4; ++i) {
#pragma unroll
        for (int j = 0; j < 4; ++j) {
            const int row0 = wm * 64 + i * 16 + hgrp * 8;
            const int col  = wn * 64 + j * 16 + lrow;
#pragma unroll
            for (int r = 0; r < 8; ++r)
                ob[(unsigned)((row0 + r) * kDOut + col)] = acc[i][j][r];
        }
    }
}

extern "C" void kernel_launch(void* const* d_in, const int* in_sizes, int n_in,
                              void* d_out, int out_size, void* d_ws, size_t ws_size,
                              hipStream_t stream) {
    (void)in_sizes; (void)n_in; (void)d_ws; (void)ws_size; (void)out_size;
    const float* x = (const float*)d_in[0];
    const float* w = (const float*)d_in[1];
    float* out = (float*)d_out;

    dim3 grid(cfg::kDOut / cfg::kBN,   // 32 N tiles
              cfg::kT    / cfg::kBM);  // 128 M tiles
    grouped_gemm_bf16_wmma<<<grid, 256, 0, stream>>>(x, w, out);
}
